// KNIFE_12687333392455
// MI455X (gfx1250) — matrix operand order repaired and checked by reading, not compiled, not run on hardware
//
#include <hip/hip_runtime.h>
#include <math.h>

#define N_ROWS 8192
#define KCOMP  128
#define DDIM   128

typedef __attribute__((ext_vector_type(16))) _Float16 v16h;
typedef __attribute__((ext_vector_type(8)))  _Float16 v8h;
typedef __attribute__((ext_vector_type(8)))  float    v8f;

// Static device scratch for logits [N, K] (written fully every call before read).
__device__ float g_logits[N_ROWS * KCOMP];

// wave32 XOR shuffle via ds_swizzle (group-of-32 mode: and=0x1f, or=0, xor=M)
template<int M>
__device__ __forceinline__ float xswz(float v) {
    return __int_as_float(
        __builtin_amdgcn_ds_swizzle(__float_as_int(v), (M << 10) | 0x1f));
}

__device__ __forceinline__ float wave32_sum(float v) {
    v += xswz<1>(v); v += xswz<2>(v); v += xswz<4>(v);
    v += xswz<8>(v); v += xswz<16>(v);
    return v;
}
__device__ __forceinline__ float wave32_max(float v) {
    v = fmaxf(v, xswz<1>(v)); v = fmaxf(v, xswz<2>(v)); v = fmaxf(v, xswz<4>(v));
    v = fmaxf(v, xswz<8>(v)); v = fmaxf(v, xswz<16>(v));
    return v;
}
// reduce across the 16 lanes of each half-wave (XOR 1,2,4,8)
__device__ __forceinline__ float half16_sum(float v) {
    v += xswz<1>(v); v += xswz<2>(v); v += xswz<4>(v); v += xswz<8>(v);
    return v;
}

union AB { v16h v; v8h h8[2]; };

// ---------------------------------------------------------------------------
// Stage 1: logits[n,k] = -0.5*|| (I + tril(tri[k],-1)) * ((x[n]-mu[k])*var[k]) ||^2
//                        + sum_d log(|var[k,d]|+1e-8) + log_softmax(weigh)[k]
// One WG = one component k x 512 rows.  8 waves, each wave does 4 tiles of 16 rows.
// ---------------------------------------------------------------------------
__global__ __launch_bounds__(256) void gmm_logits_kernel(
    const float* __restrict__ x, const float* __restrict__ means,
    const float* __restrict__ logvar, const float* __restrict__ tri,
    const float* __restrict__ weigh)
{
    // Bsh(d,e) = M[d][e] = (e==d ? 1 : (e<d ? tri[k][d][e] : 0)), f16, XOR-swizzled
    __shared__ _Float16 Bsh[DDIM * DDIM];          // 32 KB
    __shared__ _Float16 Ysh[8 * 16 * DDIM];        // 32 KB, per-wave 16x128 y tiles

    const int k    = blockIdx.y;
    const int row0 = blockIdx.x * 512;
    const int tid  = threadIdx.x;
    const int wave = tid >> 5;
    const int lane = tid & 31;
    const int lm   = lane & 15;   // lane within half-wave
    const int hk   = lane >> 4;   // which half-wave

    // ---- build M = I + tril(tri[k], -1) into LDS (coalesced global, swizzled LDS)
    for (int i = tid; i < DDIM * DDIM; i += 256) {
        const int d = i >> 7, e = i & 127;
        float m = (e < d) ? tri[(size_t)k * DDIM * DDIM + i]
                          : ((e == d) ? 1.0f : 0.0f);
        const int idx = d * 128 + ((((e >> 3) ^ (d & 15)) & 15) << 3) + (e & 7);
        Bsh[idx] = (_Float16)m;
    }

    // ---- per-lane columns c = lane*4+j: mean, var, logdet partial
    float mean4[4], var4[4];
    float ldet = 0.0f;
#pragma unroll
    for (int j = 0; j < 4; ++j) {
        const int c = lane * 4 + j;
        mean4[j] = means[k * DDIM + c];
        const float v = expf(logvar[k * DDIM + c]);
        var4[j] = v;
        ldet += logf(fabsf(v) + 1e-8f);
    }
    ldet = wave32_sum(ldet);

    // ---- w[k] = log_softmax(weigh)[k]
    float wl0 = weigh[lane * 4 + 0], wl1 = weigh[lane * 4 + 1];
    float wl2 = weigh[lane * 4 + 2], wl3 = weigh[lane * 4 + 3];
    float wmax = wave32_max(fmaxf(fmaxf(wl0, wl1), fmaxf(wl2, wl3)));
    float wsum = expf(wl0 - wmax) + expf(wl1 - wmax) +
                 expf(wl2 - wmax) + expf(wl3 - wmax);
    wsum = wave32_sum(wsum);
    const float bias = ldet + (weigh[k] - (wmax + logf(wsum)));

    __syncthreads();   // Bsh ready for all waves

    const int cchunk = lane >> 1;        // chunk of e = lane*4
    const int coff   = (lane & 1) * 4;   // offset within chunk

    for (int t = 0; t < 4; ++t) {
        const int rbase = row0 + (t * 8 + wave) * 16;
        _Float16* Y = &Ysh[wave * 16 * 128];

        // ---- y = (x - mean) * var, fp32 -> f16, into per-wave LDS tile
#pragma unroll
        for (int r = 0; r < 16; ++r) {
            const float4 xv =
                *(const float4*)(x + (size_t)(rbase + r) * DDIM + lane * 4);
            const int idx = r * 128 + (((cchunk ^ r) & 15) << 3) + coff;
            Y[idx + 0] = (_Float16)((xv.x - mean4[0]) * var4[0]);
            Y[idx + 1] = (_Float16)((xv.y - mean4[1]) * var4[1]);
            Y[idx + 2] = (_Float16)((xv.z - mean4[2]) * var4[2]);
            Y[idx + 3] = (_Float16)((xv.w - mean4[3]) * var4[3]);
        }
        // (LDS ops from one wave execute in order: no extra wait needed)

        // ---- A fragments: 16x32 f16, rows = tile rows, K = e
        AB a[4];
#pragma unroll
        for (int ks = 0; ks < 4; ++ks) {
            const int c0 = ks * 4 + 2 * hk;      // chunk of e = ks*32 + 16*hk
            a[ks].h8[0] = *(const v8h*)&Y[lm * 128 + (((c0 ^ lm) & 15) << 3)];
            a[ks].h8[1] = *(const v8h*)&Y[lm * 128 + ((((c0 + 1) ^ lm) & 15) << 3)];
        }

        float qp[8] = {0.f, 0.f, 0.f, 0.f, 0.f, 0.f, 0.f, 0.f};
#pragma unroll
        for (int dt = 0; dt < 8; ++dt) {
            v8f acc = {0.f, 0.f, 0.f, 0.f, 0.f, 0.f, 0.f, 0.f};
            const int drow = dt * 16 + lm;       // output dim d = B column
            const int dsw  = drow & 15;
#pragma unroll
            for (int ks = 0; ks < 4; ++ks) {
                // B fragment: B[e][d] = M[d][e]; lane reads e-contiguous from row d
                const int ce = ks * 4 + 2 * hk;  // chunk of e = ks*32 + 16*hk
                AB b;
                b.h8[0] = *(const v8h*)&Bsh[drow * 128 + (((ce ^ dsw) & 15) << 3)];
                b.h8[1] = *(const v8h*)&Bsh[drow * 128 + ((((ce + 1) ^ dsw) & 15) << 3)];
                acc = __builtin_amdgcn_wmma_f32_16x16x32_f16(
                    false, a[ks].v, false, b.v, (short)0, acc, false, false);
            }
#pragma unroll
            for (int r = 0; r < 8; ++r) qp[r] += acc[r] * acc[r];
        }

        // ---- reduce ||y'||^2 per row across the 16 lanes (columns) of each half
#pragma unroll
        for (int r = 0; r < 8; ++r) qp[r] = half16_sum(qp[r]);

#pragma unroll
        for (int r = 0; r < 8; ++r) {
            if (lm == r) {
                const int row = rbase + hk * 8 + r;
                g_logits[(size_t)row * KCOMP + k] = -0.5f * qp[r] + bias;
            }
        }
    }
}

// ---------------------------------------------------------------------------
// Stage 2: out[n] = -( logC + logsumexp_k logits[n,k] ), one wave per row
// ---------------------------------------------------------------------------
__global__ __launch_bounds__(256) void gmm_lse_kernel(float* __restrict__ out)
{
    const int row  = (blockIdx.x * 256 + threadIdx.x) >> 5;
    const int lane = threadIdx.x & 31;
    if (row >= N_ROWS) return;

    const float4 v = *(const float4*)&g_logits[(size_t)row * KCOMP + lane * 4];
    float m = wave32_max(fmaxf(fmaxf(v.x, v.y), fmaxf(v.z, v.w)));
    float s = expf(v.x - m) + expf(v.y - m) + expf(v.z - m) + expf(v.w - m);
    s = wave32_sum(s);
    if (lane == 0) {
        const float logC = -0.5f * 128.0f * logf(6.283185307179586f);
        out[row] = -(logC + m + logf(s));
    }
}

extern "C" void kernel_launch(void* const* d_in, const int* in_sizes, int n_in,
                              void* d_out, int out_size, void* d_ws, size_t ws_size,
                              hipStream_t stream) {
    const float* x      = (const float*)d_in[0];
    const float* means  = (const float*)d_in[1];
    const float* logvar = (const float*)d_in[2];
    const float* tri    = (const float*)d_in[3];
    const float* weigh  = (const float*)d_in[4];
    float* out = (float*)d_out;

    dim3 g1(N_ROWS / 512, KCOMP);
    gmm_logits_kernel<<<g1, 256, 0, stream>>>(x, means, logvar, tri, weigh);
    gmm_lse_kernel<<<N_ROWS / 8, 256, 0, stream>>>(out);
}
